// mymodel_53730040872986
// MI455X (gfx1250) — matrix operand order repaired and checked by reading, not compiled, not run on hardware
//
#include <hip/hip_runtime.h>
#include <hip/hip_bf16.h>

// ---------------------------------------------------------------------------
// MCCNN-style point-cloud classifier for MI455X (gfx1250, wave32, WMMA).
// f16 storage end-to-end; f32 WMMA accumulation; f32 epilogue math.
// ---------------------------------------------------------------------------

typedef __attribute__((ext_vector_type(16))) _Float16 v16h;
typedef __attribute__((ext_vector_type(8)))  _Float16 v8h;
typedef __attribute__((ext_vector_type(8)))  float    v8f;

#define LEAKY_A 0.3f
#define BATCH   16

enum { F_BIAS = 1, F_AFF = 2, F_SKIP = 4, F_LEAKY = 8, F_OUT32 = 16 };

__device__ __forceinline__ int imin(int a, int b) { return a < b ? a : b; }

__device__ __forceinline__ void epilogue(const v8f& acc, int m0, int half, int col,
                                         const float* __restrict__ bias,
                                         const float* __restrict__ g,
                                         const float* __restrict__ bt,
                                         const _Float16* __restrict__ skip,
                                         void* __restrict__ Yv, int N, int flags)
{
    const float pb = (flags & F_BIAS) ? bias[col] : 0.f;
    const float pg = (flags & F_AFF)  ? g[col]    : 1.f;
    const float pt = (flags & F_AFF)  ? bt[col]   : 0.f;
#pragma unroll
    for (int d = 0; d < 8; ++d) {
        const int r = m0 + d + (half << 3);       // D layout: row = d + 8*half
        float v = acc[d] + pb;
        v = v * pg + pt;
        if (flags & F_SKIP)  v += (float)skip[(size_t)r * N + col];
        if (flags & F_LEAKY) v = v >= 0.f ? v : LEAKY_A * v;
        if (flags & F_OUT32) ((float*)Yv)[(size_t)r * N + col] = v;
        else ((_Float16*)Yv)[(size_t)r * N + col] = (_Float16)v;
    }
}

// ---------------------------------------------------------------------------
// Pack f32 row-major W[K,N] -> f16 transposed Wt[N,K] (B operand, contiguous
// in K so B fragments are straight b128 loads).
// ---------------------------------------------------------------------------
__global__ void k_pack_wt(const float* __restrict__ W, _Float16* __restrict__ Wt,
                          int K, int N)
{
    const int idx = blockIdx.x * blockDim.x + threadIdx.x;
    if (idx >= K * N) return;
    const int n = idx / K, k = idx - n * K;
    Wt[idx] = (_Float16)W[(size_t)k * N + n];
}

__global__ void k_f32tof16(const float* __restrict__ x, _Float16* __restrict__ y,
                           int n)
{
    const int i = blockIdx.x * blockDim.x + threadIdx.x;
    if (i < n) y[i] = (_Float16)x[i];
}

// ---------------------------------------------------------------------------
// K==1 "GEMM" = outer-product broadcast: Y[r,c] = epi(X[r,0] * W[0,c]).
// Running this through the matrix pipe would be 31/32 zero padding; a plain
// VALU kernel is the right tool.
// ---------------------------------------------------------------------------
__global__ void k_outer1(const _Float16* __restrict__ X,
                         const float* __restrict__ W,   // [N] (row 0 of f32 W)
                         const float* __restrict__ bias,
                         const float* __restrict__ g,
                         const float* __restrict__ bt,
                         const _Float16* __restrict__ skip,
                         _Float16* __restrict__ Y, int M, int N, int flags)
{
    const int idx = blockIdx.x * blockDim.x + threadIdx.x;
    if (idx >= M * N) return;
    const int r = idx / N, c = idx - r * N;
    float v = (float)X[r] * W[c];
    if (flags & F_BIAS)  v += bias[c];
    if (flags & F_AFF)   v = v * g[c] + bt[c];
    if (flags & F_SKIP)  v += (float)skip[idx];
    if (flags & F_LEAKY) v = v >= 0.f ? v : LEAKY_A * v;
    Y[idx] = (_Float16)v;
}

// ---------------------------------------------------------------------------
// WMMA GEMM: Y[M,N] = epilogue( X[M,K] @ Wt[N,K]^T ), f16 storage, f32 acc.
// One wave per 16x32 C tile (one A fragment, two B fragments).
// Requires M % 16 == 0 and K % 32 == 0 (true for every call; K==1 is routed
// to k_outer1).  Pure vector loads + wmma: A = 2x b128, each B fragment is a
// contiguous v16h.  Out-of-range columns are clamped for the load and simply
// never stored (WMMA columns are independent).
// ---------------------------------------------------------------------------
__global__ __launch_bounds__(32)
void k_linear(const _Float16* __restrict__ X, const _Float16* __restrict__ Wt,
              const float* __restrict__ bias, const float* __restrict__ g,
              const float* __restrict__ bt, const _Float16* __restrict__ skip,
              void* __restrict__ Yv, int M, int N, int K, int flags)
{
    const int tilesN = (N + 31) >> 5;
    const int tm = blockIdx.x / tilesN, tn = blockIdx.x % tilesN;
    const int m0 = tm << 4, n0 = tn << 5;
    const int lane = threadIdx.x;
    const int half = lane >> 4, l16 = lane & 15;
    const int row  = m0 + l16;                    // always < M
    const int col0 = n0 + l16, col1 = col0 + 16;
    const int col0c = imin(col0, N - 1), col1c = imin(col1, N - 1);
    const _Float16* Arow = X + (size_t)row * K;
    const _Float16* B0p = Wt + (size_t)col0c * K;
    const _Float16* B1p = Wt + (size_t)col1c * K;

    v8f acc0 = {}, acc1 = {};

    for (int k0 = 0; k0 < K; k0 += 32) {
        __builtin_prefetch(Arow + k0 + 128, 0, 1);
        // A 16x32 f16, this code's k order: k = j + 16*half  -> two 8-half runs
        const v8h alo = *(const v8h*)(Arow + k0 + (half << 4));
        const v8h ahi = *(const v8h*)(Arow + k0 + 8 + (half << 4));
        v16h a;
#pragma unroll
        for (int j = 0; j < 8; ++j) { a[j] = alo[j]; a[j + 8] = ahi[j]; }
        // B 32x16 f16, same k order as A: k = j + 16*half -> 16 contiguous halves
        const v16h b0 = *(const v16h*)(B0p + k0 + (half << 4));
        const v16h b1 = *(const v16h*)(B1p + k0 + (half << 4));
        acc0 = __builtin_amdgcn_wmma_f32_16x16x32_f16(false, a, false, b0,
                                                      (short)0, acc0, false, false);
        acc1 = __builtin_amdgcn_wmma_f32_16x16x32_f16(false, a, false, b1,
                                                      (short)0, acc1, false, false);
    }

    if (col0 < N) epilogue(acc0, m0, half, col0, bias, g, bt, skip, Yv, N, flags);
    if (col1 < N) epilogue(acc1, m0, half, col1, bias, g, bt, skip, Yv, N, flags);
}

// ---------------------------------------------------------------------------
// MCConv aggregate: agg[b,o,h,f] = (1/cnt) * sum_n basis[b,o,n,h]*feat[b,n,f]
// Per (b,o): GEMM with M=H=32, K=Ni, N=F.  Basis tile (32n x 32h) computed
// into LDS as f16; feat tile (already f16) staged via 8-half vector copies;
// 8 waves cover the 2 x (F/16) output tiles.
// ---------------------------------------------------------------------------
__global__ __launch_bounds__(256)
void k_mcconv_agg(const _Float16* __restrict__ feat,  // [B,Ni,F] f16
                  const float* __restrict__ pts,      // [B,1024,3]
                  const float* __restrict__ mw1,      // [3,32]
                  const float* __restrict__ mb1,      // [32]
                  _Float16* __restrict__ agg,         // [B,No,32,F] f16
                  int Ni, int stepIn, int No, int stepOut,
                  int F, float radius)
{
    __shared__ __align__(16) _Float16 sBasis[32 * 32];
    __shared__ __align__(16) _Float16 sFeat[32 * 256];
    __shared__ float sW1[96], sB1[32];
    __shared__ float sCnt;

    const int tid = threadIdx.x;
    const int bo = blockIdx.x;
    const int b = bo / No, o = bo % No;
    const float* pb = pts + (size_t)b * 1024 * 3;
    const float ox = pb[(size_t)(o * stepOut) * 3 + 0];
    const float oy = pb[(size_t)(o * stepOut) * 3 + 1];
    const float oz = pb[(size_t)(o * stepOut) * 3 + 2];
    const float invR = 1.f / radius;

    if (tid < 96) sW1[tid] = mw1[tid];
    if (tid < 32) sB1[tid] = mb1[tid];
    if (tid == 0) sCnt = 0.f;
    __syncthreads();

    float c = 0.f;
    for (int n = tid; n < Ni; n += 256) {
        const float dx = (ox - pb[(size_t)(n * stepIn) * 3 + 0]) * invR;
        const float dy = (oy - pb[(size_t)(n * stepIn) * 3 + 1]) * invR;
        const float dz = (oz - pb[(size_t)(n * stepIn) * 3 + 2]) * invR;
        c += (dx * dx + dy * dy + dz * dz <= 1.f) ? 1.f : 0.f;
    }
    atomicAdd(&sCnt, c);
    __syncthreads();
    const float invCnt = 1.f / fmaxf(sCnt, 1.f);

    const int lane = tid & 31, wave = tid >> 5;
    const int half = lane >> 4, l16 = lane & 15;
    const int jpw = (F >> 4) * 2 >> 3;   // jobs per wave: 1 / 2 / 4
    v8f acc[4];
    {   v8f z = {};
        for (int t = 0; t < 4; ++t) acc[t] = z;  }

    const _Float16* fb = feat + (size_t)b * Ni * F;

    for (int n0 = 0; n0 < Ni; n0 += 32) {
        // --- basis tile [32n x 32h], branchless ---------------------------
        for (int idx = tid; idx < 32 * 32; idx += 256) {
            const int n = idx >> 5, h = idx & 31;
            const int nn  = n0 + n;
            const int nnc = imin(nn, Ni - 1);
            const float dx = (ox - pb[(size_t)(nnc * stepIn) * 3 + 0]) * invR;
            const float dy = (oy - pb[(size_t)(nnc * stepIn) * 3 + 1]) * invR;
            const float dz = (oz - pb[(size_t)(nnc * stepIn) * 3 + 2]) * invR;
            float v = dx * sW1[h] + dy * sW1[32 + h] + dz * sW1[64 + h] + sB1[h];
            v = v >= 0.f ? v : LEAKY_A * v;
            const bool in = (nn < Ni) && (dx * dx + dy * dy + dz * dz <= 1.f);
            sBasis[idx] = in ? (_Float16)v : (_Float16)0.f;
        }
        // --- feat tile [32n x F]: straight f16 vector copies --------------
        __builtin_prefetch(fb + (size_t)imin(n0 + 32, Ni - 1) * F, 0, 1);
        for (int idx = tid * 8; idx < 32 * F; idx += 2048) {
            const int n = idx / F;
            const int nn  = n0 + n;
            const int nnc = imin(nn, Ni - 1);
            v8h v = *(const v8h*)(fb + (size_t)nnc * F + (idx - n * F));
            if (nn >= Ni) { v8h z = {}; v = z; }
            *(v8h*)(sFeat + idx) = v;
        }
        __syncthreads();
        for (int t = 0; t < jpw; ++t) {
            const int jid = wave * jpw + t;
            const int mt = jid & 1, ft = jid >> 1;
            v16h a, bf;
#pragma unroll
            for (int j = 0; j < 16; ++j) {
                const int kn = j + (half << 4);               // A: k index (n)
                a[j]  = sBasis[kn * 32 + (mt * 16 + l16)];    // A[h][n] = basis[n][h]
                const int knb = j + (half << 4);              // B: k index (n)
                bf[j] = sFeat[knb * F + ft * 16 + l16];
            }
            acc[t] = __builtin_amdgcn_wmma_f32_16x16x32_f16(false, a, false, bf,
                                                            (short)0, acc[t],
                                                            false, false);
        }
        __syncthreads();
    }

    _Float16* ob = agg + (size_t)bo * 32 * F;
    for (int t = 0; t < jpw; ++t) {
        const int jid = wave * jpw + t;
        const int mt = jid & 1, ft = jid >> 1;
#pragma unroll
        for (int d = 0; d < 8; ++d) {
            const int h = mt * 16 + d + (half << 3);
            const int f = ft * 16 + l16;
            ob[(size_t)h * F + f] = (_Float16)(acc[t][d] * invCnt);
        }
    }
}

// ---------------------------------------------------------------------------
// Radius max-pool skip path (f16 feat)
// ---------------------------------------------------------------------------
__global__ void k_maxpool(const _Float16* __restrict__ feat,
                          const float* __restrict__ pts,
                          _Float16* __restrict__ out,
                          int Ni, int stepIn, int No, int stepOut,
                          int C, float r)
{
    const int idx = blockIdx.x * blockDim.x + threadIdx.x;
    const int total = BATCH * No * C;
    if (idx >= total) return;
    const int c = idx % C;
    const int o = (idx / C) % No;
    const int b = idx / (C * No);
    const float* pb = pts + (size_t)b * 1024 * 3;
    const float ox = pb[(size_t)(o * stepOut) * 3 + 0];
    const float oy = pb[(size_t)(o * stepOut) * 3 + 1];
    const float oz = pb[(size_t)(o * stepOut) * 3 + 2];
    const float r2 = r * r;
    const _Float16* fb = feat + (size_t)b * Ni * C + c;
    float m = -1e30f;
    for (int n = 0; n < Ni; ++n) {
        const float dx = ox - pb[(size_t)(n * stepIn) * 3 + 0];
        const float dy = oy - pb[(size_t)(n * stepIn) * 3 + 1];
        const float dz = oz - pb[(size_t)(n * stepIn) * 3 + 2];
        const float fv = (float)fb[(size_t)n * C];
        m = (dx * dx + dy * dy + dz * dz <= r2) ? fmaxf(m, fv) : m;
    }
    out[idx] = (_Float16)((m <= -5e29f) ? 0.f : m);
}

// ---------------------------------------------------------------------------
// Head: global average pool + affine + leaky (f16 in/out)
// ---------------------------------------------------------------------------
__global__ void k_mean_head(const _Float16* __restrict__ feat,
                            const float* __restrict__ g,
                            const float* __restrict__ bt,
                            _Float16* __restrict__ out, int Npts, int C)
{
    const int idx = blockIdx.x * blockDim.x + threadIdx.x;
    if (idx >= BATCH * C) return;
    const int c = idx % C, b = idx / C;
    float s = 0.f;
    for (int n = 0; n < Npts; ++n) s += (float)feat[((size_t)b * Npts + n) * C + c];
    float v = (s / (float)Npts) * g[c] + bt[c];
    out[idx] = (_Float16)(v >= 0.f ? v : LEAKY_A * v);
}

// ---------------------------------------------------------------------------
// Host orchestration
// ---------------------------------------------------------------------------
static const int NPTS_[4] = {1024, 256, 64, 16};
static const int STEP_[4] = {1, 4, 16, 64};

struct BlockP {
    const float *b1, *b2, *bs, *bt0, *bt1, *bt2, *bt3,
                *g0, *g1, *g2, *g3, *mb1, *mw1, *mw2, *w1, *w2, *ws;
};

static BlockP load_block(void* const* d_in, int base)
{
    BlockP p;
    p.b1  = (const float*)d_in[base + 0];
    p.b2  = (const float*)d_in[base + 1];
    p.bs  = (const float*)d_in[base + 2];
    p.bt0 = (const float*)d_in[base + 3];
    p.bt1 = (const float*)d_in[base + 4];
    p.bt2 = (const float*)d_in[base + 5];
    p.bt3 = (const float*)d_in[base + 6];
    p.g0  = (const float*)d_in[base + 7];
    p.g1  = (const float*)d_in[base + 8];
    p.g2  = (const float*)d_in[base + 9];
    p.g3  = (const float*)d_in[base + 10];
    p.mb1 = (const float*)d_in[base + 11];
    p.mw1 = (const float*)d_in[base + 12];
    p.mw2 = (const float*)d_in[base + 13];
    p.w1  = (const float*)d_in[base + 14];
    p.w2  = (const float*)d_in[base + 15];
    p.ws  = (const float*)d_in[base + 16];
    return p;
}

extern "C" void kernel_launch(void* const* d_in, const int* in_sizes, int n_in,
                              void* d_out, int out_size, void* d_ws, size_t ws_size,
                              hipStream_t stream)
{
    (void)out_size; (void)ws_size;

    // --- locate points / features / param base ---------------------------
    const int PTS_SZ = BATCH * 1024 * 3;   // 49152, unique among inputs
    const float *pts, *featIn0;
    int pp;  // first param-leaf index; params assumed pytree-alphabetical
    if (in_sizes[0] == PTS_SZ) {                 // insertion order: pts, feat, params
        pts = (const float*)d_in[0];
        featIn0 = (const float*)d_in[1];
        pp = 2;
    } else if (in_sizes[n_in - 1] == PTS_SZ) {   // alphabetical: feat, params, pts
        featIn0 = (const float*)d_in[0];
        pts = (const float*)d_in[n_in - 1];
        pp = 1;
    } else {                                     // fallback
        pts = (const float*)d_in[0];
        featIn0 = (const float*)d_in[1];
        pp = 2;
    }

    // conv_0, conv_1, conv_2 | d1b d1w d2b d2w hb0 hb1 hg0 hg1 | strided_0..2
    BlockP CB[3], SB[3];
    for (int i = 0; i < 3; ++i) CB[i] = load_block(d_in, pp + i * 17);
    const float* d1b = (const float*)d_in[pp + 51];
    const float* d1w = (const float*)d_in[pp + 52];
    const float* d2b = (const float*)d_in[pp + 53];
    const float* d2w = (const float*)d_in[pp + 54];
    const float* hb0 = (const float*)d_in[pp + 55];
    const float* hb1 = (const float*)d_in[pp + 56];
    const float* hg0 = (const float*)d_in[pp + 57];
    const float* hg1 = (const float*)d_in[pp + 58];
    for (int i = 0; i < 3; ++i) SB[i] = load_block(d_in, pp + 59 + i * 17);

    // --- workspace layout (f16 elements, all offsets 32B-aligned) --------
    _Float16* H = (_Float16*)d_ws;
    _Float16* FEATH  = H + 0;          //    16384
    _Float16* FEAT_A = H + 16384;      //   524288
    _Float16* FEAT_B = H + 540672;     //   524288
    _Float16* T1     = H + 1064960;    //  1048576
    _Float16* AGG    = H + 2113536;    //  8388608
    _Float16* T2     = H + 10502144;   //   262144
    _Float16* RCOMB  = H + 10764288;   //   524288
    _Float16* SKIP   = H + 11288576;   //   131072
    _Float16* HEADM  = H + 11419648;   //     8192
    _Float16* H1     = H + 11427840;   //     2048
    _Float16* WT     = H + 11429888;   //  2097152  (max K*N = 8192*256)

    auto gemm = [&](const float* Wsrc, const _Float16* X, const float* bias,
                    const float* g, const float* bt, const _Float16* skip,
                    void* Y, int M, int N, int K, int flags) {
        if (K == 1) {   // outer-product broadcast; matrix pipe is wrong tool
            k_outer1<<<(M * N + 255) / 256, 256, 0, stream>>>(
                X, Wsrc, bias, g, bt, skip, (_Float16*)Y, M, N, flags);
            return;
        }
        k_pack_wt<<<(K * N + 255) / 256, 256, 0, stream>>>(Wsrc, WT, K, N);
        const int tiles = ((M + 15) / 16) * ((N + 31) / 32);
        k_linear<<<tiles, 32, 0, stream>>>(X, WT, bias, g, bt, skip, Y,
                                           M, N, K, flags);
    };

    const float CONV_R[3] = {0.15f, 0.3f, 0.6f};
    const float POOL_R[3] = {0.1f, 0.2f, 0.4f};

    auto run_block = [&](const _Float16* fin, int liIn, int liOut, int cin, int cout,
                         bool strided, float convR, float poolR,
                         const BlockP& P, _Float16* fout) {
        const int Ni = NPTS_[liIn], No = NPTS_[liOut];
        const int sIn = STEP_[liIn], sOut = STEP_[liOut];
        const int R = cout / 2;
        // t1 = leaky(bn(feat @ w1 + b1))
        gemm(P.w1, fin, P.b1, P.g0, P.bt0, nullptr, T1,
             BATCH * Ni, R, cin, F_BIAS | F_AFF | F_LEAKY);
        // MCConv aggregate -> AGG [B,No,32,R]
        k_mcconv_agg<<<BATCH * No, 256, 0, stream>>>(T1, pts, P.mw1, P.mb1, AGG,
                                                     Ni, sIn, No, sOut, R, convR);
        // t2 = leaky(bn(AGG @ mw2))  (contraction over (h,f) = 32*R)
        gemm(P.mw2, AGG, nullptr, P.g1, P.bt1, nullptr, T2,
             BATCH * No, R, 32 * R, F_AFF | F_LEAKY);
        // rmain = bn(t2 @ w2 + b2)
        gemm(P.w2, T2, P.b2, P.g2, P.bt2, nullptr, RCOMB,
             BATCH * No, cout, R, F_BIAS | F_AFF);
        // skip path
        const _Float16* skipSrc = fin;
        if (strided) {
            const int total = BATCH * No * cin;
            k_maxpool<<<(total + 255) / 256, 256, 0, stream>>>(fin, pts, SKIP,
                                                               Ni, sIn, No, sOut,
                                                               cin, poolR);
            skipSrc = SKIP;
        }
        // out = leaky(bn(skip @ ws + bs) + rmain)
        gemm(P.ws, skipSrc, P.bs, P.g3, P.bt3, RCOMB, fout,
             BATCH * No, cout, cin, F_BIAS | F_AFF | F_SKIP | F_LEAKY);
    };

    // input features f32 -> f16
    k_f32tof16<<<(BATCH * 1024 + 255) / 256, 256, 0, stream>>>(featIn0, FEATH,
                                                               BATCH * 1024);

    run_block(FEATH,  0, 1,   1, 128, true,  CONV_R[0], POOL_R[0], SB[0], FEAT_A);
    run_block(FEAT_A, 1, 1, 128, 128, false, CONV_R[0], 0.f,       CB[0], FEAT_B);
    run_block(FEAT_B, 1, 2, 128, 256, true,  CONV_R[1], POOL_R[1], SB[1], FEAT_A);
    run_block(FEAT_A, 2, 2, 256, 256, false, CONV_R[1], 0.f,       CB[1], FEAT_B);
    run_block(FEAT_B, 2, 3, 256, 512, true,  CONV_R[2], POOL_R[2], SB[2], FEAT_A);
    run_block(FEAT_A, 3, 3, 512, 512, false, CONV_R[2], 0.f,       CB[2], FEAT_B);

    // head: mean -> bn+leaky -> d1 -> bn+leaky -> d2 (f32 out)
    k_mean_head<<<(BATCH * 512 + 255) / 256, 256, 0, stream>>>(FEAT_B, hg0, hb0,
                                                               HEADM, 16, 512);
    gemm(d1w, HEADM, d1b, hg1, hb1, nullptr, H1, BATCH, 128, 512,
         F_BIAS | F_AFF | F_LEAKY);
    gemm(d2w, H1, d2b, nullptr, nullptr, nullptr, d_out, BATCH, 40, 128,
         F_BIAS | F_OUT32);
}
